// OptimizedQuantizableGRU_7490422964599
// MI455X (gfx1250) — matrix operand order repaired and checked by reading, not compiled, not run on hardware
//
#include <hip/hip_runtime.h>
#include <hip/hip_bf16.h>
#include <stddef.h>

// Problem constants (match reference): B=32, S=512, H=1024
#define HDIM 1024
#define SEQ  512
#define BSZ  32
#define NG   3072          // 3*H (gates r,z,n stacked)
#define MROWS (BSZ * SEQ)  // 16384 rows for the batched input projection
#define RNWG 48            // workgroups in persistent recurrence kernel (48*8 waves = 384 = 2*192 tiles)

typedef __attribute__((ext_vector_type(16))) __bf16          v16bf;
typedef __attribute__((ext_vector_type(16))) unsigned short  v16us;
typedef __attribute__((ext_vector_type(8)))  unsigned short  v8us;
typedef __attribute__((ext_vector_type(8)))  float           v8f;

// ---------- bf16 split helpers (round-to-nearest-even) ----------
__device__ __forceinline__ unsigned short bf16_rne(float f) {
  unsigned u = __float_as_uint(f);
  unsigned r = u + (0x7FFFu + ((u >> 16) & 1u));
  return (unsigned short)(r >> 16);
}
__device__ __forceinline__ float bf16_up(unsigned short h) {
  return __uint_as_float(((unsigned)h) << 16);
}
__device__ __forceinline__ v16bf as_bf16(v16us v) {
  return __builtin_bit_cast(v16bf, v);
}

// ---------- WMMA operand loaders (wave32, 16x16x32 bf16) ----------
// B operand: W row-major [N, K=H]. lane l -> col n = (l&15); loads K = k0 + (l>>4)*16 + 0..15 (same order as A)
__device__ __forceinline__ v16bf wmma_ld_b(const unsigned short* __restrict__ wrows,
                                           int lane, int k0) {
  const unsigned short* p = wrows + (size_t)(lane & 15) * HDIM + k0 + ((lane >> 4) << 4);
  v8us c0 = *(const v8us*)(p);
  v8us c1 = *(const v8us*)(p + 8);
  v16us v = __builtin_shufflevector(c0, c1, 0,1,2,3,4,5,6,7,8,9,10,11,12,13,14,15);
  return as_bf16(v);
}
// A operand (pre-split bf16): lane l -> row m = (l&15); loads K = k0 + (l>>4)*16 + 0..15 (same order as B)
__device__ __forceinline__ v16bf wmma_ld_a(const unsigned short* __restrict__ arows,
                                           int lane, int k0) {
  const unsigned short* p = arows + (size_t)(lane & 15) * HDIM + k0 + ((lane >> 4) << 4);
  v8us c0 = *(const v8us*)(p);
  v8us c1 = *(const v8us*)(p + 8);
  v16us v = __builtin_shufflevector(c0, c1, 0,1,2,3,4,5,6,7,8,9,10,11,12,13,14,15);
  return as_bf16(v);
}

__device__ __forceinline__ v8f wmma_bf16(v16bf a, v16bf b, v8f c) {
  return __builtin_amdgcn_wmma_f32_16x16x32_bf16(false, a, false, b, (short)0, c, false, false);
}

// ---------- grid-wide sense-reversing barrier (persistent kernel) ----------
__device__ __forceinline__ void grid_barrier(unsigned* cnt, unsigned* gen, unsigned nwg) {
  __syncthreads();
  if (threadIdx.x == 0) {
    __threadfence();
    unsigned g = __atomic_load_n(gen, __ATOMIC_RELAXED);
    unsigned arrived = atomicAdd(cnt, 1u);
    if (arrived == nwg - 1u) {
      __atomic_store_n(cnt, 0u, __ATOMIC_RELAXED);
      __threadfence();
      atomicAdd(gen, 1u);
    } else {
      while (__atomic_load_n(gen, __ATOMIC_RELAXED) == g) {
        __builtin_amdgcn_s_sleep(1);
      }
    }
    __threadfence();
  }
  __syncthreads();
}

// ---------- elementwise f32 -> (bf16 hi, bf16 lo) split ----------
__global__ void split_bf16_kernel(const float* __restrict__ in,
                                  unsigned short* __restrict__ hi,
                                  unsigned short* __restrict__ lo, int n) {
  int stride = gridDim.x * blockDim.x;
  for (int i = blockIdx.x * blockDim.x + threadIdx.x; i < n; i += stride) {
    float x = in[i];
    unsigned short hb = bf16_rne(x);
    hi[i] = hb;
    lo[i] = bf16_rne(x - bf16_up(hb));
  }
}

// ---------- zero-init h buffers (f32 + bf16 splits) + barrier counters ----------
__global__ void init_state_kernel(float* __restrict__ h,
                                  unsigned short* __restrict__ h_hi,
                                  unsigned short* __restrict__ h_lo,
                                  unsigned* __restrict__ bar, int nh) {
  int stride = gridDim.x * blockDim.x;
  for (int i = blockIdx.x * blockDim.x + threadIdx.x; i < nh; i += stride) {
    h[i] = 0.f;
    h_hi[i] = 0;
    h_lo[i] = 0;
  }
  if (blockIdx.x == 0 && threadIdx.x < 16) bar[threadIdx.x] = 0u;
}

// ---------- batched input projection: out[M, 3H] = A[M,H] @ W[3H,H]^T + bias ----------
// split-bf16: A = Ahi+Alo, W = Whi+Wlo; out = AhiWhi + AhiWlo + AloWhi
// Each wave computes a 16(M) x 64(N) block (4 accumulators, A reused 4x).
__global__ __launch_bounds__(256) void gemm_inproj_kernel(
    const unsigned short* __restrict__ Ahi, const unsigned short* __restrict__ Alo,
    const unsigned short* __restrict__ Whi, const unsigned short* __restrict__ Wlo,
    const float* __restrict__ bias, float* __restrict__ out, int M) {
  const int lane = threadIdx.x & 31;
  const int wave = (blockIdx.x * blockDim.x + threadIdx.x) >> 5;
  const int jobsN = NG / 64;              // 48
  const int mt = wave / jobsN;
  const int n0 = (wave % jobsN) * 64;
  if (mt * 16 >= M) return;

  const unsigned short* aH = Ahi + (size_t)mt * 16 * HDIM;
  const unsigned short* aL = Alo + (size_t)mt * 16 * HDIM;

  v8f acc[4] = {};
  for (int k0 = 0; k0 < HDIM; k0 += 32) {
    v16bf ah = wmma_ld_a(aH, lane, k0);
    v16bf al = wmma_ld_a(aL, lane, k0);
#pragma unroll
    for (int s = 0; s < 4; ++s) {
      const unsigned short* wrH = Whi + (size_t)(n0 + s * 16) * HDIM;
      const unsigned short* wrL = Wlo + (size_t)(n0 + s * 16) * HDIM;
      v16bf bh = wmma_ld_b(wrH, lane, k0);
      v16bf bl = wmma_ld_b(wrL, lane, k0);
      acc[s] = wmma_bf16(ah, bh, acc[s]);
      acc[s] = wmma_bf16(al, bh, acc[s]);
      acc[s] = wmma_bf16(ah, bl, acc[s]);
    }
  }
  // C/D layout: lane n-col = lane&15 (halves), VGPR r -> row r + 8*(lane>>4)
  const int mrow = mt * 16 + ((lane >> 4) << 3);
#pragma unroll
  for (int s = 0; s < 4; ++s) {
    int n = n0 + s * 16 + (lane & 15);
    float bv = bias[n];
#pragma unroll
    for (int r = 0; r < 8; ++r) {
      out[(size_t)(mrow + r) * NG + n] = acc[s][r] + bv;
    }
  }
}

// ---------- persistent GRU recurrence (one layer) ----------
// gi: [B, S, 3H] precomputed input projection (+b_ih). h kept three ways, double-buffered:
// f32 (for z*h_prev) and bf16 hi/lo (WMMA operands, emitted by the gate stage so the
// serial GEMM chain is pure load+WMMA with no conversion VALU).
// Each wave owns one fixed 16x16 gh tile (mt in {0,1}, nt in [0,192)) for all timesteps,
// so its 16 rows of W_hh stay hot in L2. Gate stage + grid barriers per step.
__global__ __launch_bounds__(256) void gru_recurrence_kernel(
    const float* __restrict__ gi,
    const unsigned short* __restrict__ WhhHi, const unsigned short* __restrict__ WhhLo,
    const float* __restrict__ b_hh,
    float* __restrict__ h,             // [2, B, H] zero-initialized (f32)
    unsigned short* __restrict__ h_hi, // [2, B, H] bf16 hi split
    unsigned short* __restrict__ h_lo, // [2, B, H] bf16 lo split
    float* __restrict__ gh,            // [B, 3H] scratch
    unsigned* __restrict__ bar,        // 4 counters
    float* __restrict__ y_f32,         // layer 1: write f32 output here (else null)
    unsigned short* __restrict__ y_hi, unsigned short* __restrict__ y_lo, // layer 0 splits
    float* __restrict__ h_last) {      // [B, H]
  const int lane = threadIdx.x & 31;
  const int wave = (blockIdx.x << 3) | (threadIdx.x >> 5);   // 0..383
  const int nt = wave % 192;
  const int mt = wave / 192;
  const int tid = blockIdx.x * blockDim.x + threadIdx.x;

  const unsigned short* wrH = WhhHi + (size_t)nt * 16 * HDIM;
  const unsigned short* wrL = WhhLo + (size_t)nt * 16 * HDIM;

  for (int t = 0; t < SEQ; ++t) {
    const size_t curOff = (size_t)(t & 1) * (BSZ * HDIM);
    const size_t nxtOff = (size_t)((t + 1) & 1) * (BSZ * HDIM);
    const float* hcur = h + curOff;
    float* hnext      = h + nxtOff;

    // ---- stage 1: gh = h @ W_hh^T (tile per wave), pre-split bf16 operands ----
    {
      const unsigned short* aH = h_hi + curOff + (size_t)mt * 16 * HDIM;
      const unsigned short* aL = h_lo + curOff + (size_t)mt * 16 * HDIM;
      v8f acc = {};
      for (int k0 = 0; k0 < HDIM; k0 += 32) {
        v16bf ah = wmma_ld_a(aH, lane, k0);
        v16bf al = wmma_ld_a(aL, lane, k0);
        v16bf bh = wmma_ld_b(wrH, lane, k0);
        v16bf bl = wmma_ld_b(wrL, lane, k0);
        acc = wmma_bf16(ah, bh, acc);
        acc = wmma_bf16(al, bh, acc);
        acc = wmma_bf16(ah, bl, acc);
      }
      const int n = nt * 16 + (lane & 15);
      const int mrow = mt * 16 + ((lane >> 4) << 3);
#pragma unroll
      for (int r = 0; r < 8; ++r) gh[(size_t)(mrow + r) * NG + n] = acc[r];
    }
    grid_barrier(&bar[0], &bar[1], RNWG);

    // ---- stage 2: gates + h update, 32*1024 elements over all threads ----
    for (int e = tid; e < BSZ * HDIM; e += RNWG * 256) {
      const int b = e >> 10;
      const int j = e & (HDIM - 1);
      const float* git = gi + ((size_t)b * SEQ + t) * NG;
      const float* ghb = gh + (size_t)b * NG;
      float ir = git[j];
      float iz = git[HDIM + j];
      float in_ = git[2 * HDIM + j];
      float hr = ghb[j] + b_hh[j];
      float hz = ghb[HDIM + j] + b_hh[HDIM + j];
      float hn = ghb[2 * HDIM + j] + b_hh[2 * HDIM + j];
      float rg = 1.f / (1.f + __expf(-(ir + hr)));
      float zg = 1.f / (1.f + __expf(-(iz + hz)));
      float ng = tanhf(in_ + rg * hn);
      float hprev = hcur[e];
      float hnew = (1.f - zg) * ng + zg * hprev;
      hnext[e] = hnew;
      // maintain bf16 split of h for next step's WMMA stage
      unsigned short hb = bf16_rne(hnew);
      unsigned short lb = bf16_rne(hnew - bf16_up(hb));
      h_hi[nxtOff + e] = hb;
      h_lo[nxtOff + e] = lb;
      size_t yoff = ((size_t)b * SEQ + t) * HDIM + j;
      if (y_hi) {
        y_hi[yoff] = hb;
        y_lo[yoff] = lb;
      } else {
        y_f32[yoff] = hnew;
      }
      if (t == SEQ - 1) h_last[e] = hnew;
    }
    grid_barrier(&bar[2], &bar[3], RNWG);
  }
}

extern "C" void kernel_launch(void* const* d_in, const int* in_sizes, int n_in,
                              void* d_out, int out_size, void* d_ws, size_t ws_size,
                              hipStream_t stream) {
  (void)in_sizes; (void)n_in; (void)out_size; (void)ws_size;
  const float* x    = (const float*)d_in[0];
  const float* Wih0 = (const float*)d_in[1];
  const float* bih0 = (const float*)d_in[2];
  const float* Whh0 = (const float*)d_in[3];
  const float* bhh0 = (const float*)d_in[4];
  const float* Wih1 = (const float*)d_in[5];
  const float* bih1 = (const float*)d_in[6];
  const float* Whh1 = (const float*)d_in[7];
  const float* bhh1 = (const float*)d_in[8];

  float* out = (float*)d_out;                      // y1 [B,S,H] then h_n [2,B,H]
  float* y1_out  = out;
  float* hn0_out = out + (size_t)BSZ * SEQ * HDIM;
  float* hn1_out = hn0_out + (size_t)BSZ * HDIM;

  // ---- workspace layout ----
  char* ws = (char*)d_ws;
  size_t off = 0;
  auto take = [&](size_t bytes) { size_t r = off; off += (bytes + 255) & ~(size_t)255; return r; };
  unsigned* bar   = (unsigned*)(ws + take(64));
  float* hbuf     = (float*)(ws + take((size_t)2 * BSZ * HDIM * 4));
  unsigned short* hhi = (unsigned short*)(ws + take((size_t)2 * BSZ * HDIM * 2));
  unsigned short* hlo = (unsigned short*)(ws + take((size_t)2 * BSZ * HDIM * 2));
  float* ghbuf    = (float*)(ws + take((size_t)BSZ * NG * 4));
  float* gibuf    = (float*)(ws + take((size_t)MROWS * NG * 4));
  unsigned short* xhi = (unsigned short*)(ws + take((size_t)MROWS * HDIM * 2)); // reused as y0_hi
  unsigned short* xlo = (unsigned short*)(ws + take((size_t)MROWS * HDIM * 2)); // reused as y0_lo
  const size_t wsz = (size_t)NG * HDIM;
  unsigned short* wih0h = (unsigned short*)(ws + take(wsz * 2));
  unsigned short* wih0l = (unsigned short*)(ws + take(wsz * 2));
  unsigned short* whh0h = (unsigned short*)(ws + take(wsz * 2));
  unsigned short* whh0l = (unsigned short*)(ws + take(wsz * 2));
  unsigned short* wih1h = (unsigned short*)(ws + take(wsz * 2));
  unsigned short* wih1l = (unsigned short*)(ws + take(wsz * 2));
  unsigned short* whh1h = (unsigned short*)(ws + take(wsz * 2));
  unsigned short* whh1l = (unsigned short*)(ws + take(wsz * 2));

  // ---- 1) split weights + x into bf16 hi/lo ----
  split_bf16_kernel<<<2048, 256, 0, stream>>>(Wih0, wih0h, wih0l, (int)wsz);
  split_bf16_kernel<<<2048, 256, 0, stream>>>(Whh0, whh0h, whh0l, (int)wsz);
  split_bf16_kernel<<<2048, 256, 0, stream>>>(Wih1, wih1h, wih1l, (int)wsz);
  split_bf16_kernel<<<2048, 256, 0, stream>>>(Whh1, whh1h, whh1l, (int)wsz);
  split_bf16_kernel<<<8192, 256, 0, stream>>>(x, xhi, xlo, MROWS * HDIM);

  const int gemmBlocks = (MROWS / 16) * (NG / 64) / 8;   // 6144 WGs, 8 waves each

  // ---- layer 0 ----
  gemm_inproj_kernel<<<gemmBlocks, 256, 0, stream>>>(xhi, xlo, wih0h, wih0l, bih0, gibuf, MROWS);
  init_state_kernel<<<256, 256, 0, stream>>>(hbuf, hhi, hlo, bar, 2 * BSZ * HDIM);
  gru_recurrence_kernel<<<RNWG, 256, 0, stream>>>(gibuf, whh0h, whh0l, bhh0, hbuf, hhi, hlo,
                                                  ghbuf, bar, nullptr, xhi, xlo, hn0_out);

  // ---- layer 1 (y0 splits live in xhi/xlo) ----
  gemm_inproj_kernel<<<gemmBlocks, 256, 0, stream>>>(xhi, xlo, wih1h, wih1l, bih1, gibuf, MROWS);
  init_state_kernel<<<256, 256, 0, stream>>>(hbuf, hhi, hlo, bar, 2 * BSZ * HDIM);
  gru_recurrence_kernel<<<RNWG, 256, 0, stream>>>(gibuf, whh1h, whh1l, bhh1, hbuf, hhi, hlo,
                                                  ghbuf, bar, y1_out, nullptr, nullptr, hn1_out);
}
